// TransformerBlock_10110353015510
// MI455X (gfx1250) — matrix operand order repaired and checked by reading, not compiled, not run on hardware
//
#include <hip/hip_runtime.h>
#include <math.h>

// Problem constants (from reference): T=4096, B=8, H=1024, FF=4096
#define NTOK  32768      // T*B tokens
#define HDIM  1024
#define FFDIM 4096
#define LN_EPS 1e-6f
#define GELU_CC 0.7978845608028654f   // sqrt(2/pi)
#define LOG2E   1.4426950408889634f

typedef __attribute__((ext_vector_type(16))) __bf16 v16bf;
typedef __attribute__((ext_vector_type(8)))  float  v8f;

union Frag {
  v16bf v;
  uint4 q[2];
};

__device__ __forceinline__ unsigned short f32_to_bf16_rne(float f) {
  unsigned int u = __float_as_uint(f);
  u += 0x7FFFu + ((u >> 16) & 1u);     // round-to-nearest-even
  return (unsigned short)(u >> 16);
}

// Low 32 bits of a generic pointer into LDS = wave-relative LDS byte address
// (ISA: LDS aperture generic addrs are {SHARED_BASE[63:32], lds_offset[31:0]}).
__device__ __forceinline__ unsigned lds_addr32(const void* p) {
  return (unsigned)(uintptr_t)p;
}

// CDNA5 async DMA: global -> LDS, 16 bytes, tracked by ASYNCcnt.
__device__ __forceinline__ void async_copy_b128(unsigned lds, const void* g) {
  asm volatile("global_load_async_to_lds_b128 %0, %1, off"
               :: "v"(lds), "v"(g) : "memory");
}
// Async loads complete IN ORDER (ISA 08 §4.1): waiting ASYNCcnt<=4 releases the
// oldest in-flight tile (4 loads) while the newest tile's 4 loads keep running.
__device__ __forceinline__ void wait_async_le4() {
  asm volatile("s_wait_asynccnt 4" ::: "memory");
}
__device__ __forceinline__ void wait_async_le0() {
  asm volatile("s_wait_asynccnt 0" ::: "memory");
}

// ---------------------------------------------------------------------------
// fp32 -> bf16 weight conversion
// ---------------------------------------------------------------------------
__global__ __launch_bounds__(256)
void cvt_f32_bf16_kernel(const float* __restrict__ in,
                         unsigned short* __restrict__ out, int n) {
  int i = blockIdx.x * 256 + threadIdx.x;
  if (i < n) out[i] = f32_to_bf16_rne(in[i]);
}

// ---------------------------------------------------------------------------
// LayerNorm (torch variant: unbiased std (ddof=1), denom = std + eps),
// output cast to bf16 for the WMMA GEMMs. One block (256 thr) per token.
// ---------------------------------------------------------------------------
__global__ __launch_bounds__(256)
void layernorm_bf16_kernel(const float* __restrict__ x,
                           const float* __restrict__ g,
                           const float* __restrict__ b,
                           unsigned short* __restrict__ out) {
  __shared__ float red[256];
  const int row = blockIdx.x;
  const int tid = threadIdx.x;
  const float* xr = x + (size_t)row * HDIM;

  float s = 0.f;
  #pragma unroll
  for (int i = tid; i < HDIM; i += 256) s += xr[i];
  red[tid] = s; __syncthreads();
  #pragma unroll
  for (int k = 128; k > 0; k >>= 1) {
    if (tid < k) red[tid] += red[tid + k];
    __syncthreads();
  }
  const float mean = red[0] / (float)HDIM;
  __syncthreads();

  float sq = 0.f;
  #pragma unroll
  for (int i = tid; i < HDIM; i += 256) { float d = xr[i] - mean; sq += d * d; }
  red[tid] = sq; __syncthreads();
  #pragma unroll
  for (int k = 128; k > 0; k >>= 1) {
    if (tid < k) red[tid] += red[tid + k];
    __syncthreads();
  }
  const float inv = 1.0f / (sqrtf(red[0] / (float)(HDIM - 1)) + LN_EPS);

  unsigned short* orow = out + (size_t)row * HDIM;
  #pragma unroll
  for (int i = tid; i < HDIM; i += 256) {
    float v = g[i] * (xr[i] - mean) * inv + b[i];
    orow[i] = f32_to_bf16_rne(v);
  }
}

// ---------------------------------------------------------------------------
// Tiled WMMA GEMM:  D = epilogue(A @ Bw^T + bias [, resid])
//   A  : M x K, bf16 row-major  (activations)
//   Bw : N x K, bf16 row-major  (weight; GEMM is A @ Bw^T so both K-contig)
//   Block tile: 128x128, K-step 32. 256 threads = 8 waves in a 2(M)x4(N) grid.
//   Each wave: 64x32 tile = 4x2 v_wmma_f32_16x16x32_bf16 accumulators.
//   Global->LDS staging: CDNA5 async DMA (ASYNCcnt), TRIPLE-buffered so two
//   tiles are always in flight behind the WMMA work.
// MODE 0: bias -> bf16 out
// MODE 1: bias + resid -> f32 out
// MODE 2: bias + tanh-GELU -> bf16 out
// ---------------------------------------------------------------------------
#define LDSR 40   // padded LDS row stride (bf16 elems): bank-balanced b128 loads

template <int MODE>
__global__ __launch_bounds__(256)
void gemm_wmma_bf16(const unsigned short* __restrict__ A,
                    const unsigned short* __restrict__ Bw,
                    const float* __restrict__ bias,
                    const float* __restrict__ resid,
                    void* __restrict__ Dout,
                    int K, int N) {
  __shared__ unsigned short Ash[3][128 * LDSR];   // 3 x 10KB
  __shared__ unsigned short Bsh[3][128 * LDSR];   // 3 x 10KB  (60KB total)

  const int tid  = threadIdx.x;
  const int lane = tid & 31;
  const int wid  = tid >> 5;
  const int wm   = wid >> 2;          // 0..1  (M wave coord)
  const int wn   = wid & 3;           // 0..3  (N wave coord)

  const size_t m0 = (size_t)blockIdx.x * 128;
  const size_t n0 = (size_t)blockIdx.y * 128;

  // Global->LDS staging: each thread owns two 16B chunks for A and for B.
  const int r0  = tid >> 2;           // rows 0..63
  const int r1  = r0 + 64;            // rows 64..127
  const int kc0 = (tid & 3) * 8;      // bf16 elem offset in K-tile: 0/8/16/24

  const unsigned short* Aptr0 = A  + (m0 + r0) * (size_t)K + kc0;
  const unsigned short* Aptr1 = A  + (m0 + r1) * (size_t)K + kc0;
  const unsigned short* Bptr0 = Bw + (n0 + r0) * (size_t)K + kc0;
  const unsigned short* Bptr1 = Bw + (n0 + r1) * (size_t)K + kc0;

  // Per-thread LDS destinations (byte addresses) for the three buffers.
  unsigned ldsA0[3], ldsA1[3], ldsB0[3], ldsB1[3];
  #pragma unroll
  for (int s = 0; s < 3; ++s) {
    ldsA0[s] = lds_addr32(&Ash[s][r0 * LDSR + kc0]);
    ldsA1[s] = lds_addr32(&Ash[s][r1 * LDSR + kc0]);
    ldsB0[s] = lds_addr32(&Bsh[s][r0 * LDSR + kc0]);
    ldsB1[s] = lds_addr32(&Bsh[s][r1 * LDSR + kc0]);
  }

  v8f acc[4][2];
  #pragma unroll
  for (int i = 0; i < 4; ++i)
    #pragma unroll
    for (int j = 0; j < 2; ++j)
      acc[i][j] = (v8f){0.f, 0.f, 0.f, 0.f, 0.f, 0.f, 0.f, 0.f};

  const int nk = K >> 5;   // >= 32 for all our GEMMs

  // Kick off async DMA for tiles 0 and 1.
  async_copy_b128(ldsA0[0], Aptr0);
  async_copy_b128(ldsA1[0], Aptr1);
  async_copy_b128(ldsB0[0], Bptr0);
  async_copy_b128(ldsB1[0], Bptr1);
  async_copy_b128(ldsA0[1], Aptr0 + 32);
  async_copy_b128(ldsA1[1], Aptr1 + 32);
  async_copy_b128(ldsB0[1], Bptr0 + 32);
  async_copy_b128(ldsB1[1], Bptr1 + 32);

  // CDNA5 WMMA lane addressing (ISA 7.12.2):
  //  A 16x32 bf16: lane l holds row l%16; lanes 0-15 -> K chunks {0..7,16..23},
  //                lanes 16-31 -> {8..15,24..31}
  //  B 32x16 bf16: lane l holds weight-row (=N col) l%16; same K chunks as A:
  //                lanes 0-15 -> {0..7,16..23}, lanes 16-31 -> {8..15,24..31}
  const int mrow = lane & 15;
  const int aoff = (lane & 16) >> 1;  // 0 or 8
  const int boff = (lane & 16);       // 0 or 16

  int cur = 0, nxt = 1, fut = 2;
  for (int kt = 0; kt < nk; ++kt) {
    // Release the oldest tile: tile kt+1 is still outstanding iff kt+1 < nk.
    if (kt + 1 < nk) wait_async_le4();
    else             wait_async_le0();
    // Barrier: everyone's tile-kt DMA has landed, and everyone finished
    // computing on buffer `fut` (its last read was at iteration kt-1).
    __syncthreads();

    if (kt + 2 < nk) {
      const size_t ko = (size_t)(kt + 2) * 32;
      async_copy_b128(ldsA0[fut], Aptr0 + ko);
      async_copy_b128(ldsA1[fut], Aptr1 + ko);
      async_copy_b128(ldsB0[fut], Bptr0 + ko);
      async_copy_b128(ldsB1[fut], Bptr1 + ko);
    }
    if (kt + 3 < nk) {
      __builtin_prefetch(Aptr0 + (size_t)(kt + 3) * 32, 0, 1);  // global_prefetch_b8
      __builtin_prefetch(Bptr0 + (size_t)(kt + 3) * 32, 0, 1);
    }

    Frag afr[4], bfr[2];
    #pragma unroll
    for (int j = 0; j < 2; ++j) {
      const unsigned short* base = &Bsh[cur][(wn * 32 + j * 16 + mrow) * LDSR];
      bfr[j].q[0] = *(const uint4*)(base + aoff);
      bfr[j].q[1] = *(const uint4*)(base + aoff + 16);
    }
    #pragma unroll
    for (int i = 0; i < 4; ++i) {
      const unsigned short* base = &Ash[cur][(wm * 64 + i * 16 + mrow) * LDSR];
      afr[i].q[0] = *(const uint4*)(base + aoff);
      afr[i].q[1] = *(const uint4*)(base + aoff + 16);
    }

    #pragma unroll
    for (int i = 0; i < 4; ++i)
      #pragma unroll
      for (int j = 0; j < 2; ++j)
        acc[i][j] = __builtin_amdgcn_wmma_f32_16x16x32_bf16(
            false, afr[i].v, false, bfr[j].v, (short)0, acc[i][j], false, false);

    const int t = cur; cur = nxt; nxt = fut; fut = t;
  }

  // Epilogue. D layout (ISA 7.12.2): vgpr r, lanes 0-15 -> M=r, N=lane;
  // lanes 16-31 -> M=8+r, N=lane-16.
  #pragma unroll
  for (int j = 0; j < 2; ++j) {
    const size_t col = n0 + wn * 32 + j * 16 + mrow;
    const float bi = bias[col];
    #pragma unroll
    for (int i = 0; i < 4; ++i) {
      const size_t mbase = m0 + wm * 64 + i * 16 + aoff;
      #pragma unroll
      for (int r = 0; r < 8; ++r) {
        const size_t idx = (mbase + r) * (size_t)N + col;
        float val = acc[i][j][r] + bi;
        if (MODE == 2) {
          // gelu_tanh(x) = x * sigmoid(2*c*(x + 0.044715 x^3))
          const float z2 = 2.0f * GELU_CC * (val + 0.044715f * val * val * val);
          const float e  = __builtin_amdgcn_exp2f(-LOG2E * z2);   // exp(-2z)
          val = val * __builtin_amdgcn_rcpf(1.0f + e);
        }
        if (MODE == 1) {
          ((float*)Dout)[idx] = resid[idx] + val;
        } else {
          ((unsigned short*)Dout)[idx] = f32_to_bf16_rne(val);
        }
      }
    }
  }
}

// ---------------------------------------------------------------------------
// Host pipeline. seq_len==1 attention => softmax==1 => o == v, so q/k are
// mathematically dead:  attn_out = (LN1(x) @ wv^T + bv) @ wo^T + bo.
// ---------------------------------------------------------------------------
extern "C" void kernel_launch(void* const* d_in, const int* in_sizes, int n_in,
                              void* d_out, int out_size, void* d_ws, size_t ws_size,
                              hipStream_t stream) {
  (void)in_sizes; (void)n_in; (void)out_size; (void)ws_size;

  const float* x   = (const float*)d_in[0];
  const float* wv  = (const float*)d_in[5];
  const float* bv  = (const float*)d_in[6];
  const float* wo  = (const float*)d_in[7];
  const float* bo  = (const float*)d_in[8];
  const float* w1  = (const float*)d_in[9];
  const float* b1  = (const float*)d_in[10];
  const float* w2  = (const float*)d_in[11];
  const float* b2  = (const float*)d_in[12];
  const float* g1  = (const float*)d_in[13];
  const float* be1 = (const float*)d_in[14];
  const float* g2  = (const float*)d_in[15];
  const float* be2 = (const float*)d_in[16];
  float* out = (float*)d_out;

  // Workspace carve-out (256B aligned)
  char* ws = (char*)d_ws;
  size_t off = 0;
  auto carve = [&](size_t bytes) -> char* {
    char* p = ws + off;
    off = (off + bytes + 255) & ~(size_t)255;
    return p;
  };
  unsigned short* wvb = (unsigned short*)carve((size_t)HDIM * HDIM * 2);
  unsigned short* wob = (unsigned short*)carve((size_t)HDIM * HDIM * 2);
  unsigned short* w1b = (unsigned short*)carve((size_t)FFDIM * HDIM * 2);
  unsigned short* w2b = (unsigned short*)carve((size_t)FFDIM * HDIM * 2);
  unsigned short* hb  = (unsigned short*)carve((size_t)NTOK * HDIM * 2);
  unsigned short* vb  = (unsigned short*)carve((size_t)NTOK * HDIM * 2);
  float*          x1  = (float*)         carve((size_t)NTOK * HDIM * 4);
  unsigned short* h2b = (unsigned short*)carve((size_t)NTOK * HDIM * 2);
  unsigned short* ub  = (unsigned short*)carve((size_t)NTOK * FFDIM * 2);

  // 1) Weights -> bf16
  {
    const int nHH = HDIM * HDIM;
    const int nFH = FFDIM * HDIM;
    cvt_f32_bf16_kernel<<<nHH / 256, 256, 0, stream>>>(wv, wvb, nHH);
    cvt_f32_bf16_kernel<<<nHH / 256, 256, 0, stream>>>(wo, wob, nHH);
    cvt_f32_bf16_kernel<<<nFH / 256, 256, 0, stream>>>(w1, w1b, nFH);
    cvt_f32_bf16_kernel<<<nFH / 256, 256, 0, stream>>>(w2, w2b, nFH);
  }

  const dim3 blk(256);
  const dim3 gHH(NTOK / 128, HDIM / 128);    // 256 x 8
  const dim3 gHF(NTOK / 128, FFDIM / 128);   // 256 x 32

  // 2) h = LN1(x)  (bf16)
  layernorm_bf16_kernel<<<NTOK, blk, 0, stream>>>(x, g1, be1, hb);

  // 3) v = h @ wv^T + bv  (bf16)
  gemm_wmma_bf16<0><<<gHH, blk, 0, stream>>>(hb, wvb, bv, nullptr, vb, HDIM, HDIM);

  // 4) x1 = x + (v @ wo^T + bo)  (f32)
  gemm_wmma_bf16<1><<<gHH, blk, 0, stream>>>(vb, wob, bo, x, x1, HDIM, HDIM);

  // 5) h2 = LN2(x1)  (bf16)
  layernorm_bf16_kernel<<<NTOK, blk, 0, stream>>>(x1, g2, be2, h2b);

  // 6) u = gelu(h2 @ w1^T + b1)  (bf16)
  gemm_wmma_bf16<2><<<gHF, blk, 0, stream>>>(h2b, w1b, b1, nullptr, ub, HDIM, FFDIM);

  // 7) out = x1 + (u @ w2^T + b2)  (f32)
  gemm_wmma_bf16<1><<<gHH, blk, 0, stream>>>(ub, w2b, b2, x1, out, FFDIM, HDIM);
}
